// PaiDGCNN2_77429670412674
// MI455X (gfx1250) — matrix operand order repaired and checked by reading, not compiled, not run on hardware
//
#include <hip/hip_runtime.h>
#include <hip/hip_bf16.h>
#include <math.h>
#include <stdint.h>

// ---------------------------------------------------------------------------
// DGCNN (PaiDGCNN2) forward, fused for MI455X (gfx1250, wave32, WMMA).
//  - dgconv fully fused per point: gather -> F(LDS,bf16) -> WMMA GEMM1 ->
//    softmax(k) -> gate -> G -> VALU GEMM2 -> BN -> GELU.  The [B,N,k,2C]
//    edge tensor (335MB) never reaches HBM.
//  - conv5 + BN + GELU + max/mean pool fused; h5 (67MB) never materialized.
//    Activation tiles staged via GLOBAL_LOAD_ASYNC_TO_LDS_B128 into a
//    double-buffered LDS ring (ASYNCcnt / s_wait_asynccnt), shared by all
//    4 waves of the block; B-fragment ds loads software-pipelined one step
//    ahead of the WMMA chain so LDS latency hides behind the matrix op.
//  - bf16 inputs / f32 accumulate via v_wmma_f32_16x16x32_bf16.
//  - KNN kept in f32 VALU (top-k ordering is precision sensitive).
// Workspace: ~52 MB (hcat f32 + hcatT bf16 + idx + pooled + z1 + z2).
// ---------------------------------------------------------------------------

#define DEVINL __device__ __forceinline__

typedef __attribute__((ext_vector_type(16))) __bf16 v16bf;
typedef __attribute__((ext_vector_type(8)))  __bf16 v8bf;
typedef __attribute__((ext_vector_type(8)))  float  v8f;

constexpr int BATCH = 8;
constexpr int NPTS  = 2048;
constexpr int KNNK  = 20;
constexpr float BN_EPS = 1e-5f;

DEVINL v8f wmma_bf16(v16bf a, v16bf b, v8f c) {
  // D = A(16x32 bf16) * B(32x16 bf16) + C(16x16 f32)
  return __builtin_amdgcn_wmma_f32_16x16x32_bf16(false, a, false, b, (short)0, c,
                                                 false, false);
}

// A-fragment K index for element e (0..15) of v16bf, per ISA 7.12.2
// (16-bit A 16x32): lanes 0-15: V0..3 -> K0..7, V4..7 -> K16..23;
// lanes 16-31: K8..15, K24..31.
DEVINL int kmapA(int e, int half) {
  int v = e >> 1;
  return ((v & 4) ? 16 : 0) + (half ? 8 : 0) + ((v & 3) << 1) + (e & 1);
}
// conv5 A-frags (same order as its LDS B-frags): lane half 0 holds K0..15, half 1 holds K16..31, contiguous.
DEVINL int kmapB(int e, int half) { return (half ? 16 : 0) + e; }

DEVINL float gelu_erf(float x) {
  return 0.5f * x * (1.0f + erff(x * 0.70710678118654752f));
}

// async 16B copy global -> LDS (tracked by ASYNCcnt). lds_off is the wave-
// relative LDS byte offset = low 32 bits of the flat shared-memory address.
DEVINL void async_g2l_b128(unsigned lds_off, const void* gptr) {
  asm volatile("global_load_async_to_lds_b128 %0, %1, off"
               :: "v"(lds_off), "v"((unsigned long long)(uintptr_t)gptr)
               : "memory");
}
DEVINL void wait_asynccnt_8() {
  asm volatile("s_wait_asynccnt 0x8" ::: "memory");
}
DEVINL void wait_asynccnt_0() {
  asm volatile("s_wait_asynccnt 0x0" ::: "memory");
}

// ---------------------------------------------------------------------------
// KNN: one wave per (b, m). LDS caches -||x_m - x_j||^2 for all j, then 20
// masked-argmax passes with cross-lane shfl reduction (ties -> lowest index,
// matching jax.lax.top_k).
// ---------------------------------------------------------------------------
template<int CIN>
__global__ void __launch_bounds__(128)
knn_kernel(const float* __restrict__ xin, long bstride, int* __restrict__ idx) {
  __shared__ float dist[4][NPTS];
  __shared__ float cent[4][CIN];
  const int t = threadIdx.x & 31;
  const int w = threadIdx.x >> 5;
  const int m = blockIdx.x * 4 + w;
  const int b = blockIdx.y;
  const float* xb = xin + (long)b * bstride;

  for (int c = t; c < CIN; c += 32) cent[w][c] = xb[(long)c * NPTS + m];
  __syncthreads();

  for (int j = t; j < NPTS; j += 32) {
    float acc = 0.f;
    #pragma unroll 4
    for (int c = 0; c < CIN; ++c) {
      float d = cent[w][c] - xb[(long)c * NPTS + j];
      acc -= d * d;
    }
    dist[w][j] = acc;
  }
  __syncthreads();

  int* out = idx + ((long)b * NPTS + m) * KNNK;
  for (int p = 0; p < KNNK; ++p) {
    float bv = -3.0e38f; int bi = NPTS;
    for (int j = t; j < NPTS; j += 32) {
      float vv = dist[w][j];
      if (vv > bv || (vv == bv && j < bi)) { bv = vv; bi = j; }
    }
    #pragma unroll
    for (int off = 16; off >= 1; off >>= 1) {
      float ov = __shfl_xor(bv, off);
      int   oi = __shfl_xor(bi, off);
      if (ov > bv || (ov == bv && oi < bi)) { bv = ov; bi = oi; }
    }
    if (t == 0) { out[p] = bi; dist[w][bi] = -3.3e38f; }
    __syncthreads();
  }
}

// ---------------------------------------------------------------------------
// Fused dgconv: one wave per point. F [20 x 2C] in LDS (bf16, zero padded to
// 32 x C2P). GEMM1 via WMMA per 16-column chunk, softmax over k per column,
// gated reduction into G, then VALU GEMM2 + BN + GELU. Writes f32 channel-
// major slice (next-layer knn/gather) and bf16 n-major slice (conv5 B-frags).
// ---------------------------------------------------------------------------
template<int C2, int O>
__global__ void __launch_bounds__(64)
dgconv_kernel(const float* __restrict__ xin, long bstride,
              const int* __restrict__ idx,
              const float* __restrict__ W1, const float* __restrict__ W2,
              const float* __restrict__ bng, const float* __restrict__ bnb,
              const float* __restrict__ bnm, const float* __restrict__ bnv,
              float* __restrict__ hcat_out,   // already + chOff*NPTS
              __bf16* __restrict__ hT_out) {  // already + chOff
  constexpr int C2P = (C2 < 32) ? 32 : C2;
  constexpr int CIN = C2 / 2;
  constexpr int NC  = C2P / 16;
  constexpr int NK  = C2P / 32;

  __shared__ __bf16 Fsh[2][32][C2P];
  __shared__ float  Ssh[2][32][16];
  __shared__ float  Gsh[2][C2P];

  const int t = threadIdx.x & 31;
  const int w = threadIdx.x >> 5;
  const int n = blockIdx.x * 2 + w;
  const int b = blockIdx.y;
  const float* xb = xin + (long)b * bstride;
  const int* ir = idx + ((long)b * NPTS + n) * KNNK;

  // zero-pad F (rows >= 20 and cols >= C2 must be 0)
  for (int e = t; e < 32 * C2P; e += 32) (&Fsh[w][0][0])[e] = (__bf16)0.f;
  __syncthreads();

  // gather + edge features: [diff | center]
  for (int e = t; e < KNNK * CIN; e += 32) {
    int kk = e / CIN, c = e % CIN;
    int j = ir[kk];
    float xc = xb[(long)c * NPTS + n];
    float xj = xb[(long)c * NPTS + j];
    Fsh[w][kk][c]       = (__bf16)(xj - xc);
    Fsh[w][kk][CIN + c] = (__bf16)xc;
  }
  __syncthreads();

  const int half = t >> 4, lr = t & 15;

  for (int cc = 0; cc < NC; ++cc) {
    v8f acc0 = {}; v8f acc1 = {};
    for (int kt = 0; kt < NK; ++kt) {
      v16bf a0, a1, bb;
      const int cp = cc * 16 + lr;
      #pragma unroll
      for (int e = 0; e < 16; ++e) {
        int ka = kt * 32 + kmapA(e, half);
        a0[e] = Fsh[w][lr][ka];
        a1[e] = Fsh[w][16 + lr][ka];
        int kb = kt * 32 + kmapA(e, half);
        bb[e] = (cp < C2 && kb < C2) ? (__bf16)W1[(long)cp * C2 + kb]
                                     : (__bf16)0.f;
      }
      acc0 = wmma_bf16(a0, bb, acc0);
      acc1 = wmma_bf16(a1, bb, acc1);
    }
    // scatter S tile (D layout: lane -> col lr, VGPR r -> row r + 8*half)
    #pragma unroll
    for (int r = 0; r < 8; ++r) {
      int row = r + (half ? 8 : 0);
      Ssh[w][row][lr]      = acc0[r];
      Ssh[w][16 + row][lr] = acc1[r];
    }
    __syncthreads();

    if (half == 0) {
      int cp = cc * 16 + lr;
      if (cp < C2) {
        float mx = -3.0e38f;
        #pragma unroll
        for (int kk = 0; kk < KNNK; ++kk) mx = fmaxf(mx, Ssh[w][kk][lr]);
        float ex[KNNK];
        float se = 0.f;
        #pragma unroll
        for (int kk = 0; kk < KNNK; ++kk) {
          ex[kk] = __expf(Ssh[w][kk][lr] - mx);
          se += ex[kk];
        }
        float ga = 0.f;
        #pragma unroll
        for (int kk = 0; kk < KNNK; ++kk)
          ga += ex[kk] * (float)Fsh[w][kk][cp];
        Gsh[w][cp] = ga / se;
      }
    }
    __syncthreads();
  }

  // GEMM2 (M=1 per wave) + BN + GELU + dual-layout store
  for (int o = t; o < O; o += 32) {
    float acc = 0.f;
    const float* w2r = W2 + (long)o * C2;
    #pragma unroll 4
    for (int c = 0; c < C2; ++c) acc += Gsh[w][c] * w2r[c];
    float s = bng[o] / sqrtf(bnv[o] + BN_EPS);
    float val = gelu_erf((acc - bnm[o]) * s + bnb[o]);
    hcat_out[(long)b * 512 * NPTS + (long)o * NPTS + n] = val;
    hT_out[((long)b * NPTS + n) * 512 + o] = (__bf16)val;
  }
}

// ---------------------------------------------------------------------------
// conv5 (512 -> 1024) + BN + GELU + global max/mean pool, fused.
// Block = 4 waves, one (b, 16-channel tile) per wave. All 4 waves consume the
// SAME activation tile (16 points x 512 ch bf16 = 16KB, contiguous), so it is
// staged once per block via async global->LDS copies, double buffered:
// tile t+1 streams in (ASYNCcnt) while the 16-WMMA chain runs on tile t.
// LDS rows padded 16B so the 16 B-frag lanes hit disjoint bank stripes.
// B-fragment ds loads are pipelined one K-step ahead of the WMMA they feed.
// ---------------------------------------------------------------------------
constexpr int C5PAD = 520;  // 512 + 8 bf16 -> row stride 1040B (16B aligned)

__global__ void __launch_bounds__(128)
conv5_pool_kernel(const __bf16* __restrict__ hT, const float* __restrict__ W5,
                  const float* __restrict__ g, const float* __restrict__ bb,
                  const float* __restrict__ m, const float* __restrict__ v,
                  float* __restrict__ pooled) {
  __shared__ __bf16 bbuf[2][16][C5PAD];

  const int tid = threadIdx.x;
  const int t = tid & 31;
  const int w = tid >> 5;
  const int otile = blockIdx.x * 4 + w;
  const int b = blockIdx.y;
  const int half = t >> 4, lr = t & 15;
  const int obase = otile * 16;

  // weights -> A fragments (registers), invariant over point tiles
  v16bf afrag[16];
  const float* wrow = W5 + (long)(obase + lr) * 512;
  #pragma unroll
  for (int kt = 0; kt < 16; ++kt) {
    #pragma unroll
    for (int e = 0; e < 16; ++e)
      afrag[kt][e] = (__bf16)wrow[kt * 32 + kmapB(e, half)];
  }

  float sc[8], sh[8];
  #pragma unroll
  for (int r = 0; r < 8; ++r) {
    int ch = obase + r + (half ? 8 : 0);
    float s = g[ch] / sqrtf(v[ch] + BN_EPS);
    sc[r] = s; sh[r] = bb[ch] - m[ch] * s;
  }

  float vmax[8], vsum[8];
  #pragma unroll
  for (int r = 0; r < 8; ++r) { vmax[r] = -3.0e38f; vsum[r] = 0.f; }

  const char* hb = (const char*)(hT + (long)b * NPTS * 512);  // bytes

  // issue async copies of point-tile `nt` into buffer `buf`
  auto issue_tile = [&](int nt, int buf) {
    const char* gbase = hb + (long)nt * 16 * 1024;  // 16 rows x 1024B
    #pragma unroll
    for (int i = 0; i < 8; ++i) {
      int id = i * 128 + tid;        // 1024 chunks of 16B
      int r = id >> 6, c = id & 63;
      unsigned lds = (unsigned)(uintptr_t)(const void*)&bbuf[buf][r][c * 8];
      async_g2l_b128(lds, gbase + r * 1024 + c * 16);
    }
  };

  issue_tile(0, 0);

  for (int nt = 0; nt < NPTS / 16; ++nt) {
    const int cur = nt & 1;
    if (nt + 1 < NPTS / 16) {
      issue_tile(nt + 1, cur ^ 1);
      wait_asynccnt_8();   // tile nt complete (async loads retire in order)
    } else {
      wait_asynccnt_0();
    }
    __syncthreads();       // publish LDS tile to all 4 waves

    // 16-step WMMA chain; B fragments pipelined one step ahead so the
    // ds_load latency hides behind the previous WMMA.
    const __bf16* pbase = &bbuf[cur][lr][half * 16];
    v8bf lo = *reinterpret_cast<const v8bf*>(pbase);
    v8bf hi = *reinterpret_cast<const v8bf*>(pbase + 8);
    v8f acc = {};
    #pragma unroll
    for (int kt = 0; kt < 16; ++kt) {
      v8bf lo2 = {}, hi2 = {};
      if (kt + 1 < 16) {
        lo2 = *reinterpret_cast<const v8bf*>(pbase + (kt + 1) * 32);
        hi2 = *reinterpret_cast<const v8bf*>(pbase + (kt + 1) * 32 + 8);
      }
      v16bf bbv;
      #pragma unroll
      for (int e = 0; e < 8; ++e) { bbv[e] = lo[e]; bbv[8 + e] = hi[e]; }
      acc = wmma_bf16(afrag[kt], bbv, acc);
      lo = lo2; hi = hi2;
    }
    #pragma unroll
    for (int r = 0; r < 8; ++r) {
      float val = gelu_erf(acc[r] * sc[r] + sh[r]);
      vmax[r] = fmaxf(vmax[r], val);
      vsum[r] += val;
    }
    __syncthreads();       // all waves done with `cur` before it is reused
  }

  // reduce over the 16 lanes (columns) of each half-wave
  #pragma unroll
  for (int r = 0; r < 8; ++r) {
    #pragma unroll
    for (int off = 8; off >= 1; off >>= 1) {
      vmax[r] = fmaxf(vmax[r], __shfl_xor(vmax[r], off));
      vsum[r] += __shfl_xor(vsum[r], off);
    }
  }
  if (lr == 0) {
    #pragma unroll
    for (int r = 0; r < 8; ++r) {
      int ch = obase + r + (half ? 8 : 0);
      pooled[(long)b * 2048 + ch]        = vmax[r];
      pooled[(long)b * 2048 + 1024 + ch] = vsum[r] * (1.f / NPTS);
    }
  }
}

// ---------------------------------------------------------------------------
// Head MLP (tiny: 8 rows)
// ---------------------------------------------------------------------------
__global__ void lin_bn_gelu_kernel(const float* __restrict__ in,
                                   const float* __restrict__ W,
                                   const float* __restrict__ bias,
                                   const float* __restrict__ g,
                                   const float* __restrict__ bb,
                                   const float* __restrict__ m,
                                   const float* __restrict__ v,
                                   float* __restrict__ out, int IN, int OUT) {
  int i = blockIdx.x * blockDim.x + threadIdx.x;
  if (i >= BATCH * OUT) return;
  int r = i / OUT, o = i % OUT;
  const float* wr = W + (long)o * IN;
  const float* xr = in + (long)r * IN;
  float acc = bias ? bias[o] : 0.f;
  for (int k = 0; k < IN; ++k) acc += xr[k] * wr[k];
  float s = g[o] / sqrtf(v[o] + BN_EPS);
  out[i] = gelu_erf((acc - m[o]) * s + bb[o]);
}

__global__ void lin_kernel(const float* __restrict__ in,
                           const float* __restrict__ W,
                           const float* __restrict__ bias,
                           float* __restrict__ out, int IN, int OUT) {
  int i = blockIdx.x * blockDim.x + threadIdx.x;
  if (i >= BATCH * OUT) return;
  int r = i / OUT, o = i % OUT;
  const float* wr = W + (long)o * IN;
  const float* xr = in + (long)r * IN;
  float acc = bias[o];
  for (int k = 0; k < IN; ++k) acc += xr[k] * wr[k];
  out[i] = acc;
}

// ---------------------------------------------------------------------------
extern "C" void kernel_launch(void* const* d_in, const int* in_sizes, int n_in,
                              void* d_out, int out_size, void* d_ws, size_t ws_size,
                              hipStream_t stream) {
  (void)in_sizes; (void)n_in; (void)out_size; (void)ws_size;
  const float* x       = (const float*)d_in[0];
  const float* w1[4]   = {(const float*)d_in[1], (const float*)d_in[3],
                          (const float*)d_in[5], (const float*)d_in[7]};
  const float* w2[4]   = {(const float*)d_in[2], (const float*)d_in[4],
                          (const float*)d_in[6], (const float*)d_in[8]};
  const float* conv5_w = (const float*)d_in[9];
  const float *bng[7], *bnb[7], *bnm[7], *bnv[7];
  for (int i = 0; i < 7; ++i) {
    int base = 10 + 4 * i;
    bng[i] = (const float*)d_in[base + 0];
    bnb[i] = (const float*)d_in[base + 1];
    bnm[i] = (const float*)d_in[base + 2];
    bnv[i] = (const float*)d_in[base + 3];
  }
  const float* lin1_w = (const float*)d_in[38];
  const float* lin2_w = (const float*)d_in[39];
  const float* lin2_b = (const float*)d_in[40];
  const float* lin3_w = (const float*)d_in[41];
  const float* lin3_b = (const float*)d_in[42];

  // workspace layout
  char* ws = (char*)d_ws;
  float*  hcat   = (float*)ws;                                   // [8,512,2048] f32
  __bf16* hT     = (__bf16*)(ws + 33554432);                     // [8,2048,512] bf16
  int*    idx    = (int*)(ws + 33554432 + 16777216);             // [8,2048,20]
  float*  pooled = (float*)(ws + 33554432 + 16777216 + 1310720); // [8,2048]
  float*  z1     = pooled + BATCH * 2048;                        // [8,512]
  float*  z2     = z1 + BATCH * 512;                             // [8,256]

  dim3 gk(NPTS / 4, BATCH), bk(128);
  dim3 gd(NPTS / 2, BATCH), bd(64);
  const long BS_X = 3L * NPTS, BS_H = 512L * NPTS;

  // layer 1: C=3 -> 64
  knn_kernel<3><<<gk, bk, 0, stream>>>(x, BS_X, idx);
  dgconv_kernel<6, 64><<<gd, bd, 0, stream>>>(x, BS_X, idx, w1[0], w2[0],
      bng[0], bnb[0], bnm[0], bnv[0], hcat + 0L * NPTS, hT + 0);
  // layer 2: 64 -> 64 (input h1 @ ch 0)
  knn_kernel<64><<<gk, bk, 0, stream>>>(hcat, BS_H, idx);
  dgconv_kernel<128, 64><<<gd, bd, 0, stream>>>(hcat, BS_H, idx, w1[1], w2[1],
      bng[1], bnb[1], bnm[1], bnv[1], hcat + 64L * NPTS, hT + 64);
  // layer 3: 64 -> 128 (input h2 @ ch 64)
  knn_kernel<64><<<gk, bk, 0, stream>>>(hcat + 64L * NPTS, BS_H, idx);
  dgconv_kernel<128, 128><<<gd, bd, 0, stream>>>(hcat + 64L * NPTS, BS_H, idx,
      w1[2], w2[2], bng[2], bnb[2], bnm[2], bnv[2],
      hcat + 128L * NPTS, hT + 128);
  // layer 4: 128 -> 256 (input h3 @ ch 128)
  knn_kernel<128><<<gk, bk, 0, stream>>>(hcat + 128L * NPTS, BS_H, idx);
  dgconv_kernel<256, 256><<<gd, bd, 0, stream>>>(hcat + 128L * NPTS, BS_H, idx,
      w1[3], w2[3], bng[3], bnb[3], bnm[3], bnv[3],
      hcat + 256L * NPTS, hT + 256);

  // conv5 + bn5 + gelu + max/mean pool (async LDS pipeline)
  conv5_pool_kernel<<<dim3(16, BATCH), 128, 0, stream>>>(
      hT, conv5_w, bng[4], bnb[4], bnm[4], bnv[4], pooled);

  // head
  lin_bn_gelu_kernel<<<(BATCH * 512 + 255) / 256, 256, 0, stream>>>(
      pooled, lin1_w, nullptr, bng[5], bnb[5], bnm[5], bnv[5], z1, 2048, 512);
  lin_bn_gelu_kernel<<<(BATCH * 256 + 255) / 256, 256, 0, stream>>>(
      z1, lin2_w, lin2_b, bng[6], bnb[6], bnm[6], bnv[6], z2, 512, 256);
  lin_kernel<<<(BATCH * 40 + 255) / 256, 256, 0, stream>>>(
      z2, lin3_w, lin3_b, (float*)d_out, 256, 40);
}
